// SelfAttentionModel_52097953300852
// MI455X (gfx1250) — matrix operand order repaired and checked by reading, not compiled, not run on hardware
//
#include <hip/hip_runtime.h>

// ---------------- problem constants ----------------
#define N_NODES   100000
#define N_EDGES   1600000
#define IN_DIM    128
#define OUT_DIM   128
#define NUM_HEADS 8
#define QKV_COLS  384          // Q|K|V concatenated

typedef __attribute__((ext_vector_type(16))) _Float16 v16h;
typedef __attribute__((ext_vector_type(8)))  float    v8f;

union AFrag { v16h v; int4 q[2]; };
union BFrag { v16h v; int4 q[2]; };

// ---------------- workspace layout (bytes) ----------------
static constexpr size_t XH_BYTES  = (size_t)N_NODES * IN_DIM * 2;          // 25,600,000
static constexpr size_t WH_BYTES  = (size_t)QKV_COLS * IN_DIM * 2;         // 98,304
static constexpr size_t BC_BYTES  = (size_t)QKV_COLS * 4;                  // 1,536
static constexpr size_t QKV_BYTES = (size_t)N_NODES * QKV_COLS * 4;        // 153,600,000
static constexpr size_t SC_BYTES  = (size_t)N_EDGES * NUM_HEADS * 4;       // 51,200,000

static constexpr size_t OFF_XH   = 0;
static constexpr size_t OFF_WH   = OFF_XH + XH_BYTES;
static constexpr size_t OFF_BC   = OFF_WH + WH_BYTES;
static constexpr size_t OFF_QKV  = OFF_BC + BC_BYTES;
static constexpr size_t OFF_SC   = OFF_QKV + QKV_BYTES;
static constexpr size_t OFF_SSUM = OFF_SC + SC_BYTES;

// Relaxed device-scope FP32 atomic add: lowers to the native non-returning
// global_atomic_add_f32 (L2 atomic unit) rather than a CAS loop.
__device__ __forceinline__ void atomic_add_f32_agent(float* addr, float val) {
    __hip_atomic_fetch_add(addr, val, __ATOMIC_RELAXED, __HIP_MEMORY_SCOPE_AGENT);
}

// ---------------- utility kernels ----------------
__global__ void zero_f4(float4* __restrict__ p, int n4) {
    int i = blockIdx.x * blockDim.x + threadIdx.x;
    if (i < n4) p[i] = make_float4(0.f, 0.f, 0.f, 0.f);
}

// convert x (fp32, [N,128]) -> f16
__global__ void cvt_x_f16(const float* __restrict__ x, _Float16* __restrict__ xh, int n4) {
    int i = blockIdx.x * blockDim.x + threadIdx.x;
    if (i >= n4) return;
    float4 f = ((const float4*)x)[i];
    _Float16* o = xh + (size_t)i * 4;
    o[0] = (_Float16)f.x; o[1] = (_Float16)f.y;
    o[2] = (_Float16)f.z; o[3] = (_Float16)f.w;
}

// pack Wq|Wk|Wv -> Wh f16 [384,128], bq|bk|bv -> bcat f32 [384]
__global__ void pack_weights(const float* __restrict__ Wq, const float* __restrict__ Wk,
                             const float* __restrict__ Wv, const float* __restrict__ bq,
                             const float* __restrict__ bk, const float* __restrict__ bv,
                             _Float16* __restrict__ Wh, float* __restrict__ bcat) {
    int i = blockIdx.x * blockDim.x + threadIdx.x;
    const int WSZ = OUT_DIM * IN_DIM;               // 16384
    if (i < QKV_COLS * IN_DIM) {
        int r = i >> 7;                              // output column 0..383
        float w = (r < 128) ? Wq[i] : (r < 256) ? Wk[i - WSZ] : Wv[i - 2 * WSZ];
        Wh[i] = (_Float16)w;
    }
    if (i < QKV_COLS) {
        bcat[i] = (i < 128) ? bq[i] : (i < 256) ? bk[i - 128] : bv[i - 256];
    }
}

// ---------------- fused QKV GEMM: QKV[N,384] = x @ [Wq|Wk|Wv]^T + b ----------------
// grid = (6250, 3), block = 256 (8 waves). Wave w of block (bx,by) computes the
// 16x16 tile at rows bx*16, cols by*128 + w*16. K=128 in four WMMA k-steps of 32.
__global__ __launch_bounds__(256) void qkv_gemm_wmma(
        const _Float16* __restrict__ xh, const _Float16* __restrict__ Wh,
        const float* __restrict__ bcat, float* __restrict__ QKV) {
    const int lane = threadIdx.x & 31;
    const int wave = threadIdx.x >> 5;
    const int rowBase = blockIdx.x * 16;
    const int colBase = blockIdx.y * 128 + wave * 16;

    const int m  = lane & 15;     // A row / B col handled by this lane
    const int hi = lane >> 4;     // lane-half select

    // A: lane holds row m, halves K = hi*8 + {0..7} and {16..23}  (ISA 16-bit A 16x32 layout)
    const _Float16* aptr = xh + (size_t)(rowBase + m) * IN_DIM + hi * 8;
    // B: lane holds col m, halves K = hi*8 + {0..7} and {16..23} (matches A)
    const _Float16* bptr = Wh + (size_t)(colBase + m) * IN_DIM + hi * 8;

    v8f acc = {};
    #pragma unroll
    for (int kt = 0; kt < 4; ++kt) {
        const int k0 = kt * 32;
        AFrag a; BFrag b;
        a.q[0] = *(const int4*)(aptr + k0);        // K = k0+hi*8 .. +7
        a.q[1] = *(const int4*)(aptr + k0 + 16);   // K = k0+hi*8+16 .. +23
        b.q[0] = *(const int4*)(bptr + k0);        // K = k0+hi*8 .. +7
        b.q[1] = *(const int4*)(bptr + k0 + 16);   // K = k0+hi*8+16 .. +23
        acc = __builtin_amdgcn_wmma_f32_16x16x32_f16(
                  false, a.v, false, b.v, (short)0, acc, false, false);
    }

    // D layout: VGPR r -> row 8*hi + r (lanes 0-15: M=0..7, lanes 16-31: M=8..15), col = lane&15
    const int nn = lane & 15;
    const float bias = bcat[colBase + nn];
    #pragma unroll
    for (int r = 0; r < 8; ++r) {
        const int mm = hi * 8 + r;
        QKV[(size_t)(rowBase + mm) * QKV_COLS + colBase + nn] = acc[r] + bias;
    }
}

// ---------------- per-edge scores + per-destination score sums ----------------
// one thread per (edge, head): 8 consecutive lanes read one node's contiguous 512B Q/K rows
__global__ __launch_bounds__(256) void edge_scores(
        const int* __restrict__ src, const int* __restrict__ dst,
        const float* __restrict__ QKV, float* __restrict__ score,
        float* __restrict__ ssum) {
    int tid = blockIdx.x * blockDim.x + threadIdx.x;
    if (tid >= N_EDGES * NUM_HEADS) return;
    const int e = tid >> 3;
    const int h = tid & 7;
    const int s = src[e];
    const int d = dst[e];
    const float4* q = (const float4*)(QKV + (size_t)s * QKV_COLS + h * 16);        // Q block
    const float4* k = (const float4*)(QKV + (size_t)d * QKV_COLS + 128 + h * 16);  // K block
    float acc = 0.f;
    #pragma unroll
    for (int i = 0; i < 4; ++i) {
        float4 qa = q[i], kb = k[i];
        acc += qa.x * kb.x + qa.y * kb.y + qa.z * kb.z + qa.w * kb.w;
    }
    score[tid] = acc;
    atomic_add_f32_agent(&ssum[(size_t)d * NUM_HEADS + h], acc);
}

// ---------------- weighted aggregation: out[dst] += V[src] * alpha ----------------
// one thread per (edge, 4 channels): 32 lanes cover one edge's 128 channels
__global__ __launch_bounds__(256) void edge_aggregate(
        const int* __restrict__ src, const int* __restrict__ dst,
        const float* __restrict__ QKV, const float* __restrict__ score,
        const float* __restrict__ ssum, float* __restrict__ out) {
    int tid = blockIdx.x * blockDim.x + threadIdx.x;
    if (tid >= N_EDGES * 32) return;
    const int e = tid >> 5;
    const int c = (tid & 31) * 4;     // channel base 0..124
    const int h = c >> 4;             // head of this 4-channel slice
    const int s = src[e];
    const int d = dst[e];
    const float alpha = score[(size_t)e * NUM_HEADS + h] *
                        (1.0f / ssum[(size_t)d * NUM_HEADS + h]);
    const float4 v = *(const float4*)(QKV + (size_t)s * QKV_COLS + 256 + c);       // V block
    float* o = out + (size_t)d * OUT_DIM + c;
    atomic_add_f32_agent(o + 0, v.x * alpha);
    atomic_add_f32_agent(o + 1, v.y * alpha);
    atomic_add_f32_agent(o + 2, v.z * alpha);
    atomic_add_f32_agent(o + 3, v.w * alpha);
}

// ---------------- host launcher ----------------
extern "C" void kernel_launch(void* const* d_in, const int* in_sizes, int n_in,
                              void* d_out, int out_size, void* d_ws, size_t ws_size,
                              hipStream_t stream) {
    const float* x  = (const float*)d_in[0];
    const int*   src = (const int*)d_in[1];
    const int*   dst = (const int*)d_in[2];
    const float* Wq = (const float*)d_in[3];
    const float* bq = (const float*)d_in[4];
    const float* Wk = (const float*)d_in[5];
    const float* bk = (const float*)d_in[6];
    const float* Wv = (const float*)d_in[7];
    const float* bv = (const float*)d_in[8];
    float* out = (float*)d_out;

    char* ws = (char*)d_ws;
    _Float16* xh   = (_Float16*)(ws + OFF_XH);
    _Float16* Wh   = (_Float16*)(ws + OFF_WH);
    float*    bcat = (float*)(ws + OFF_BC);
    float*    QKV  = (float*)(ws + OFF_QKV);
    float*    score= (float*)(ws + OFF_SC);
    float*    ssum = (float*)(ws + OFF_SSUM);

    const int B = 256;

    // x -> f16 (3.2M float4 threads)
    {
        int n4 = N_NODES * IN_DIM / 4;
        cvt_x_f16<<<(n4 + B - 1) / B, B, 0, stream>>>(x, xh, n4);
    }
    // weights/bias pack
    pack_weights<<<(QKV_COLS * IN_DIM + B - 1) / B, B, 0, stream>>>(Wq, Wk, Wv, bq, bk, bv, Wh, bcat);
    // zero score sums and output accumulator
    {
        int n4 = N_NODES * NUM_HEADS / 4;
        zero_f4<<<(n4 + B - 1) / B, B, 0, stream>>>((float4*)ssum, n4);
    }
    {
        int n4 = N_NODES * OUT_DIM / 4;
        zero_f4<<<(n4 + B - 1) / B, B, 0, stream>>>((float4*)out, n4);
    }
    // fused QKV GEMM via WMMA: rows 100000/16 = 6250 tiles, cols 384 = 3 groups of 128
    {
        dim3 grid(N_NODES / 16, QKV_COLS / 128);
        qkv_gemm_wmma<<<grid, B, 0, stream>>>(xh, Wh, bcat, QKV);
    }
    // per-edge scores + destination sums
    {
        int n = N_EDGES * NUM_HEADS;
        edge_scores<<<(n + B - 1) / B, B, 0, stream>>>(src, dst, QKV, score, ssum);
    }
    // weighted scatter into output
    {
        int n = N_EDGES * 32;
        edge_aggregate<<<(n + B - 1) / B, B, 0, stream>>>(src, dst, QKV, score, ssum, out);
    }
}
